// PianoSVSep_47485158425285
// MI455X (gfx1250) — matrix operand order repaired and checked by reading, not compiled, not run on hardware
//
#include <hip/hip_runtime.h>
#include <math.h>

#define NN   100000
#define FIN  64
#define HH   128
#define LLAY 2
#define RREL 3
#define EE   800000
#define EPOT 500000
#define ECH  200000

typedef __attribute__((ext_vector_type(16))) _Float16 v16h;
typedef __attribute__((ext_vector_type(8)))  _Float16 v8h;
typedef __attribute__((ext_vector_type(8)))  float    v8f;

#define WMMA16(ACC, A, B) \
  (ACC) = __builtin_amdgcn_wmma_f32_16x16x32_f16(false, (A), false, (B), (short)0, (ACC), false, false)

// ---- WMMA operand loaders (layouts per CDNA5 ISA 7.12.2) ----
// A (16x32 f16): lane l holds row M=l&15; elems 0..7 = K[k0+koffA .. +7],
// elems 8..15 = K[k0+16+koffA .. +7], koffA = (l<16)?0:8.
__device__ __forceinline__ v16h ld_a_h16(const _Float16* p) {
  v8h lo = *reinterpret_cast<const v8h*>(p);
  v8h hi = *reinterpret_cast<const v8h*>(p + 16);
  v16h a;
#pragma unroll
  for (int i = 0; i < 8; ++i) { a[i] = lo[i]; a[i + 8] = hi[i]; }
  return a;
}
__device__ __forceinline__ v16h ld_a_f32(const float* p, float scale) {
  v16h a;
#pragma unroll
  for (int i = 0; i < 8; ++i) a[i] = (_Float16)(p[i] * scale);
#pragma unroll
  for (int i = 0; i < 8; ++i) a[i + 8] = (_Float16)(p[i + 16] * scale);
  return a;
}
// B (32x16, B[k][n] = W[n][k]): lane l holds col N=l&15; elems 0..7 = K[k0+koffB .. +7],
// elems 8..15 = K[k0+16+koffB .. +7], koffB = (l<16)?0:8.
__device__ __forceinline__ v16h ld_b(const _Float16* p) {
  v8h lo = *reinterpret_cast<const v8h*>(p);
  v8h hi = *reinterpret_cast<const v8h*>(p + 16);
  v16h b;
#pragma unroll
  for (int i = 0; i < 8; ++i) { b[i] = lo[i]; b[i + 8] = hi[i]; }
  return b;
}

// ---- per-row LayerNorm over a 16x128 LDS tile (one wave, 32 lanes) ----
__device__ __forceinline__ void row_ln_inplace(float (*tile)[132], float* mu, float* rs,
                                               const float* __restrict__ g,
                                               const float* __restrict__ b, int lane) {
  int m = lane & 15, half = lane >> 4;
  float s = 0.f;
  for (int c = half * 64; c < half * 64 + 64; ++c) s += tile[m][c];
  s += __shfl_xor(s, 16, 32);
  float mean = s * (1.f / 128.f);
  float vs = 0.f;
  for (int c = half * 64; c < half * 64 + 64; ++c) { float d = tile[m][c] - mean; vs += d * d; }
  vs += __shfl_xor(vs, 16, 32);
  if (half == 0) { mu[m] = mean; rs[m] = rsqrtf(vs * (1.f / 128.f) + 1e-5f); }
  __syncthreads();
  for (int row = 0; row < 16; ++row) {
    float mm = mu[row], rr = rs[row];
    for (int c = lane; c < 128; c += 32)
      tile[row][c] = (tile[row][c] - mm) * rr * g[c] + b[c];
  }
  __syncthreads();
}

// ---- utility kernels ----
__global__ void zero_f32_kernel(float* __restrict__ p, size_t n) {
  size_t i = (size_t)blockIdx.x * blockDim.x + threadIdx.x;
  if (i < n) p[i] = 0.f;
}

__global__ void wcvt_kernel(const float* __restrict__ src, _Float16* __restrict__ dst,
                            int rows, int src_ld, int cols) {
  int i = blockIdx.x * blockDim.x + threadIdx.x;
  if (i >= rows * cols) return;
  int r = i / cols, c = i - r * cols;
  dst[i] = (_Float16)src[(size_t)r * src_ld + c];
}

// ---- frontend: h = LN(relu(x @ Win^T + b_in)) ----
__global__ void frontend_kernel(const float* __restrict__ x, const _Float16* __restrict__ Win16,
                                const float* __restrict__ b_in, const float* __restrict__ lng,
                                const float* __restrict__ lnb,
                                float* __restrict__ hf32, _Float16* __restrict__ h16) {
  __shared__ float tile[16][132];
  __shared__ float mu[16], rs[16];
  int lane = threadIdx.x;
  int m0 = blockIdx.x * 16;
  int m = lane & 15, half = lane >> 4;
  int koffA = half * 8, koffB = half * 8;
  v8f acc[8];
#pragma unroll
  for (int t = 0; t < 8; ++t)
#pragma unroll
    for (int i = 0; i < 8; ++i) acc[t][i] = 0.f;
  for (int k0 = 0; k0 < FIN; k0 += 32) {
    v16h a = ld_a_f32(x + (size_t)(m0 + m) * FIN + k0 + koffA, 1.0f);
#pragma unroll
    for (int t = 0; t < 8; ++t) {
      v16h b = ld_b(Win16 + (size_t)(t * 16 + m) * FIN + k0 + koffB);
      WMMA16(acc[t], a, b);
    }
  }
#pragma unroll
  for (int t = 0; t < 8; ++t)
#pragma unroll
    for (int v = 0; v < 8; ++v) {
      int row = v + half * 8, col = t * 16 + m;
      tile[row][col] = fmaxf(acc[t][v] + b_in[col], 0.f);
    }
  __syncthreads();
  row_ln_inplace(tile, mu, rs, lng, lnb, lane);
  for (int row = 0; row < 16; ++row)
    for (int c = lane; c < 128; c += 32) {
      float v = tile[row][c];
      size_t idx = (size_t)(m0 + row) * HH + c;
      hf32[idx] = v;
      h16[idx] = (_Float16)v;
    }
}

// ---- SAGE aggregation scatter: agg[dst] += h[src]; cnt[dst]++ ----
__global__ void scatter_kernel(const int* __restrict__ ei, const float* __restrict__ hf32,
                               float* __restrict__ agg, int* __restrict__ cnt) {
  int tid = blockIdx.x * blockDim.x + threadIdx.x;
  int e = tid >> 5;
  if (e >= EE) return;
  int lane = tid & 31;
  int src = ei[e];
  int dst = ei[EE + e];
  const float4 v = *reinterpret_cast<const float4*>(hf32 + (size_t)src * HH + lane * 4);
  float* d = agg + (size_t)dst * HH + lane * 4;
  atomicAdd(d + 0, v.x);
  atomicAdd(d + 1, v.y);
  atomicAdd(d + 2, v.z);
  atomicAdd(d + 3, v.w);
  if (lane == 0) atomicAdd(cnt + dst, 1);
}

// ---- per-relation GEMM: acc (+)= (agg/cnt) @ Wl^T + bl + h @ Wr^T ----
__global__ void sage_gemm_kernel(const float* __restrict__ agg, const int* __restrict__ cnt,
                                 const _Float16* __restrict__ h16,
                                 const _Float16* __restrict__ Wl16,
                                 const _Float16* __restrict__ Wr16,
                                 const float* __restrict__ bl,
                                 float* __restrict__ accout, int first) {
  int lane = threadIdx.x;
  int m0 = blockIdx.x * 16;
  int m = lane & 15, half = lane >> 4;
  int koffA = half * 8, koffB = half * 8;
  int grow = m0 + m;
  float invc = 1.f / fmaxf((float)cnt[grow], 1.f);
  v8f acc[8];
#pragma unroll
  for (int t = 0; t < 8; ++t)
#pragma unroll
    for (int i = 0; i < 8; ++i) acc[t][i] = 0.f;
  for (int k0 = 0; k0 < HH; k0 += 32) {
    v16h a1 = ld_a_f32(agg + (size_t)grow * HH + k0 + koffA, invc);
    v16h a2 = ld_a_h16(h16 + (size_t)grow * HH + k0 + koffA);
#pragma unroll
    for (int t = 0; t < 8; ++t) {
      v16h b1 = ld_b(Wl16 + (size_t)(t * 16 + m) * HH + k0 + koffB);
      WMMA16(acc[t], a1, b1);
      v16h b2 = ld_b(Wr16 + (size_t)(t * 16 + m) * HH + k0 + koffB);
      WMMA16(acc[t], a2, b2);
    }
  }
#pragma unroll
  for (int t = 0; t < 8; ++t)
#pragma unroll
    for (int v = 0; v < 8; ++v) {
      int row = v + half * 8, col = t * 16 + m;
      size_t idx = (size_t)(m0 + row) * HH + col;
      float val = acc[t][v] + bl[col];
      if (first) accout[idx] = val;
      else accout[idx] += val;
    }
}

// ---- layer-0 finalize: acc = relu(acc/3); accumulate per-feature sum, sumsq ----
__global__ void finalize0_kernel(float* __restrict__ acc, float* __restrict__ colsum,
                                 float* __restrict__ colsq) {
  int c = threadIdx.x;  // 128
  int r0 = blockIdx.x * 32;
  float s = 0.f, sq = 0.f;
  for (int i = 0; i < 32; ++i) {
    int r = r0 + i;
    if (r >= NN) break;
    size_t idx = (size_t)r * HH + c;
    float v = fmaxf(acc[idx] * (1.f / 3.f), 0.f);
    acc[idx] = v;
    s += v;
    sq += v * v;
  }
  atomicAdd(colsum + c, s);
  atomicAdd(colsq + c, sq);
}

__global__ void gn_finish_kernel(const float* __restrict__ colsum, const float* __restrict__ colsq,
                                 const float* __restrict__ gw, const float* __restrict__ ga,
                                 float* __restrict__ shiftb, float* __restrict__ mulb) {
  int c = threadIdx.x;
  float mean = colsum[c] * (1.f / (float)NN);
  float ex2 = colsq[c] * (1.f / (float)NN);
  float a = ga[c];
  float var = ex2 - 2.f * a * mean * mean + a * a * mean * mean;
  shiftb[c] = a * mean;
  mulb[c] = gw[c] * rsqrtf(var + 1e-5f);
}

__global__ void gn_apply_kernel(const float* __restrict__ acc, const float* __restrict__ shiftb,
                                const float* __restrict__ mulb, const float* __restrict__ gb,
                                float* __restrict__ hf32, _Float16* __restrict__ h16) {
  size_t i = (size_t)blockIdx.x * blockDim.x + threadIdx.x;
  if (i >= (size_t)NN * HH) return;
  int c = (int)(i & 127);
  float v = (acc[i] - shiftb[c]) * mulb[c] + gb[c];
  hf32[i] = v;
  h16[i] = (_Float16)v;
}

__global__ void finalize_last_kernel(const float* __restrict__ acc, float* __restrict__ hout,
                                     _Float16* __restrict__ h16) {
  size_t i = (size_t)blockIdx.x * blockDim.x + threadIdx.x;
  if (i >= (size_t)NN * HH) return;
  float v = acc[i] * (1.f / 3.f);
  hout[i] = v;
  h16[i] = (_Float16)v;
}

// ---- staff head: sigmoid(LN(relu(h@W1^T+b1)) @ W2^T + b2) ----
__global__ void staff_kernel(const _Float16* __restrict__ h16, const _Float16* __restrict__ W116,
                             const float* __restrict__ b1, const float* __restrict__ lng,
                             const float* __restrict__ lnb, const float* __restrict__ W2,
                             const float* __restrict__ b2, float* __restrict__ out) {
  __shared__ float tile[16][132];
  __shared__ float mu[16], rs[16];
  int lane = threadIdx.x;
  int m0 = blockIdx.x * 16;
  int m = lane & 15, half = lane >> 4;
  int koffA = half * 8, koffB = half * 8;
  v8f acc[8];
#pragma unroll
  for (int t = 0; t < 8; ++t)
#pragma unroll
    for (int i = 0; i < 8; ++i) acc[t][i] = 0.f;
  for (int k0 = 0; k0 < HH; k0 += 32) {
    v16h a = ld_a_h16(h16 + (size_t)(m0 + m) * HH + k0 + koffA);
#pragma unroll
    for (int t = 0; t < 8; ++t) {
      v16h b = ld_b(W116 + (size_t)(t * 16 + m) * HH + k0 + koffB);
      WMMA16(acc[t], a, b);
    }
  }
#pragma unroll
  for (int t = 0; t < 8; ++t)
#pragma unroll
    for (int v = 0; v < 8; ++v) {
      int row = v + half * 8, col = t * 16 + m;
      tile[row][col] = fmaxf(acc[t][v] + b1[col], 0.f);
    }
  __syncthreads();
  row_ln_inplace(tile, mu, rs, lng, lnb, lane);
  // lane -> (row = m, output j = half); st_W2 is (2,128)
  float s = 0.f;
  for (int c = 0; c < 128; ++c) s += tile[m][c] * W2[half * 128 + c];
  s += b2[half];
  out[(size_t)(m0 + m) * 2 + half] = 1.f / (1.f + expf(-s));
}

// ---- chord head (gathered K=256 GEMM) ----
__global__ void chord_kernel(const int* __restrict__ pce, const _Float16* __restrict__ h16,
                             const _Float16* __restrict__ W116, const float* __restrict__ b1,
                             const float* __restrict__ lng, const float* __restrict__ lnb,
                             const float* __restrict__ W2, const float* __restrict__ b2s,
                             float* __restrict__ out) {
  __shared__ float tile[16][132];
  __shared__ float mu[16], rs[16];
  int lane = threadIdx.x;
  int e0 = blockIdx.x * 16;
  int m = lane & 15, half = lane >> 4;
  int e = e0 + m;
  int cs = pce[e], cd = pce[ECH + e];
  v8f acc[8];
#pragma unroll
  for (int t = 0; t < 8; ++t)
#pragma unroll
    for (int i = 0; i < 8; ++i) acc[t][i] = 0.f;
  for (int k0 = 0; k0 < 256; k0 += 32) {
    const _Float16* abase = (k0 < 128) ? (h16 + (size_t)cs * HH + k0)
                                       : (h16 + (size_t)cd * HH + (k0 - 128));
    v16h a = ld_a_h16(abase + half * 8);
#pragma unroll
    for (int t = 0; t < 8; ++t) {
      v16h b = ld_b(W116 + (size_t)(t * 16 + m) * 256 + k0 + half * 8);
      WMMA16(acc[t], a, b);
    }
  }
#pragma unroll
  for (int t = 0; t < 8; ++t)
#pragma unroll
    for (int v = 0; v < 8; ++v) {
      int row = v + half * 8, col = t * 16 + m;
      tile[row][col] = fmaxf(acc[t][v] + b1[col], 0.f);
    }
  __syncthreads();
  row_ln_inplace(tile, mu, rs, lng, lnb, lane);
  float s = 0.f;
  for (int c = half * 64; c < half * 64 + 64; ++c) s += tile[m][c] * W2[c];
  s += __shfl_xor(s, 16, 32);
  if (half == 0) out[e] = s + b2s[0];
}

// ---- edge decoder head: K=256 GEMM + rank-3 scalar-feature correction ----
__global__ void decoder_kernel(const int* __restrict__ pe, const _Float16* __restrict__ h16,
                               const _Float16* __restrict__ W116, const float* __restrict__ W1f,
                               const float* __restrict__ b1, const float* __restrict__ lng,
                               const float* __restrict__ lnb, const float* __restrict__ W2,
                               const float* __restrict__ b2s,
                               const float* __restrict__ onsets, const float* __restrict__ durations,
                               const float* __restrict__ pitches, const float* __restrict__ onset_beat,
                               const float* __restrict__ duration_beat, const float* __restrict__ ts_beats,
                               float* __restrict__ out) {
  __shared__ float tile[16][132];
  __shared__ float mu[16], rs[16];
  __shared__ float sosc[16], soh[16], spsc[16];
  int lane = threadIdx.x;
  int e0 = blockIdx.x * 16;
  int m = lane & 15, half = lane >> 4;
  int e = e0 + m;
  int nrow = pe[e], ncol = pe[EPOT + e];
  if (half == 0) {
    float offb = onset_beat[nrow] + duration_beat[nrow];
    float nd = onset_beat[ncol] - offb;
    sosc[m] = 1.f - tanhf(nd / ts_beats[ncol]);
    soh[m] = (onsets[ncol] == (onsets[nrow] + durations[nrow])) ? 1.f : 0.f;
    spsc[m] = fabsf(pitches[ncol] - pitches[nrow]) * (1.f / 127.f);
  }
  v8f acc[8];
#pragma unroll
  for (int t = 0; t < 8; ++t)
#pragma unroll
    for (int i = 0; i < 8; ++i) acc[t][i] = 0.f;
  for (int k0 = 0; k0 < 256; k0 += 32) {
    const _Float16* abase = (k0 < 128) ? (h16 + (size_t)nrow * HH + k0)
                                       : (h16 + (size_t)ncol * HH + (k0 - 128));
    v16h a = ld_a_h16(abase + half * 8);
#pragma unroll
    for (int t = 0; t < 8; ++t) {
      v16h b = ld_b(W116 + (size_t)(t * 16 + m) * 256 + k0 + half * 8);
      WMMA16(acc[t], a, b);
    }
  }
  __syncthreads();
#pragma unroll
  for (int t = 0; t < 8; ++t) {
    int col = t * 16 + m;
    float w256 = W1f[(size_t)col * 259 + 256];
    float w257 = W1f[(size_t)col * 259 + 257];
    float w258 = W1f[(size_t)col * 259 + 258];
    float bb = b1[col];
#pragma unroll
    for (int v = 0; v < 8; ++v) {
      int row = v + half * 8;
      float val = acc[t][v] + bb + sosc[row] * w256 + soh[row] * w257 + spsc[row] * w258;
      tile[row][col] = fmaxf(val, 0.f);
    }
  }
  __syncthreads();
  row_ln_inplace(tile, mu, rs, lng, lnb, lane);
  float s = 0.f;
  for (int c = half * 64; c < half * 64 + 64; ++c) s += tile[m][c] * W2[c];
  s += __shfl_xor(s, 16, 32);
  if (half == 0) out[e] = s + b2s[0];
}

extern "C" void kernel_launch(void* const* d_in, const int* in_sizes, int n_in,
                              void* d_out, int out_size, void* d_ws, size_t ws_size,
                              hipStream_t stream) {
  const float* x           = (const float*)d_in[0];
  const int*   edge_index  = (const int*)d_in[1];
  const int*   pot_edges   = (const int*)d_in[2];
  const int*   pot_chord   = (const int*)d_in[3];
  const float* onsets      = (const float*)d_in[5];
  const float* durations   = (const float*)d_in[6];
  const float* pitches     = (const float*)d_in[7];
  const float* onset_beat  = (const float*)d_in[8];
  const float* durat_beat  = (const float*)d_in[9];
  const float* ts_beats    = (const float*)d_in[10];
  const float* Win   = (const float*)d_in[11];
  const float* b_in  = (const float*)d_in[12];
  const float* lnin_g = (const float*)d_in[13];
  const float* lnin_b = (const float*)d_in[14];
  const float* Wl    = (const float*)d_in[15];
  const float* bl    = (const float*)d_in[16];
  const float* Wr    = (const float*)d_in[17];
  const float* gn_w  = (const float*)d_in[18];
  const float* gn_b  = (const float*)d_in[19];
  const float* gn_a  = (const float*)d_in[20];
  const float* cp_W1 = (const float*)d_in[21];
  const float* cp_b1 = (const float*)d_in[22];
  const float* cp_lng = (const float*)d_in[23];
  const float* cp_lnb = (const float*)d_in[24];
  const float* cp_W2 = (const float*)d_in[25];
  const float* cp_b2 = (const float*)d_in[26];
  const float* st_W1 = (const float*)d_in[27];
  const float* st_b1 = (const float*)d_in[28];
  const float* st_lng = (const float*)d_in[29];
  const float* st_lnb = (const float*)d_in[30];
  const float* st_W2 = (const float*)d_in[31];
  const float* st_b2 = (const float*)d_in[32];
  const float* d_W1  = (const float*)d_in[33];
  const float* d_b1  = (const float*)d_in[34];
  const float* d_lng = (const float*)d_in[35];
  const float* d_lnb = (const float*)d_in[36];
  const float* d_W2  = (const float*)d_in[37];
  const float* d_b2  = (const float*)d_in[38];

  float* outp = (float*)d_out;
  float* out_edge  = outp;                                    // EPOT
  float* out_staff = outp + EPOT;                             // 2*NN
  float* out_h     = outp + EPOT + 2 * NN;                    // 128*NN
  float* out_pool  = outp + EPOT + 2 * NN + (size_t)HH * NN;  // ECH

  // ---- workspace layout ----
  char* base = (char*)d_ws;
  size_t off = 0;
  auto alloc = [&](size_t bytes) { size_t o = off; off += (bytes + 255) & ~(size_t)255; return o; };
  _Float16* h16   = (_Float16*)(base + alloc((size_t)NN * HH * 2));
  float*    hf32  = (float*)   (base + alloc((size_t)NN * HH * 4));
  float*    accb  = (float*)   (base + alloc((size_t)NN * HH * 4));
  float*    agg   = (float*)   (base + alloc((size_t)NN * HH * 4));
  int*      cnt   = (int*)     (base + alloc((size_t)NN * 4));
  float*    colsum = (float*)  (base + alloc(128 * 4));
  float*    colsq  = (float*)  (base + alloc(128 * 4));
  float*    shiftb = (float*)  (base + alloc(128 * 4));
  float*    mulb   = (float*)  (base + alloc(128 * 4));
  _Float16* Win16  = (_Float16*)(base + alloc((size_t)HH * FIN * 2));
  _Float16* Wl16   = (_Float16*)(base + alloc((size_t)LLAY * RREL * HH * HH * 2));
  _Float16* Wr16   = (_Float16*)(base + alloc((size_t)LLAY * RREL * HH * HH * 2));
  _Float16* st16   = (_Float16*)(base + alloc((size_t)HH * HH * 2));
  _Float16* cp16   = (_Float16*)(base + alloc((size_t)HH * 256 * 2));
  _Float16* dd16   = (_Float16*)(base + alloc((size_t)HH * 256 * 2));

  // ---- weight conversion to f16 (packed for WMMA B feeds) ----
  wcvt_kernel<<<(HH * FIN + 255) / 256, 256, 0, stream>>>(Win, Win16, HH, FIN, FIN);
  wcvt_kernel<<<(LLAY * RREL * HH * HH + 255) / 256, 256, 0, stream>>>(Wl, Wl16, LLAY * RREL * HH, HH, HH);
  wcvt_kernel<<<(LLAY * RREL * HH * HH + 255) / 256, 256, 0, stream>>>(Wr, Wr16, LLAY * RREL * HH, HH, HH);
  wcvt_kernel<<<(HH * HH + 255) / 256, 256, 0, stream>>>(st_W1, st16, HH, HH, HH);
  wcvt_kernel<<<(HH * 256 + 255) / 256, 256, 0, stream>>>(cp_W1, cp16, HH, 256, 256);
  wcvt_kernel<<<(HH * 256 + 255) / 256, 256, 0, stream>>>(d_W1, dd16, HH, 259, 256);

  // ---- frontend ----
  frontend_kernel<<<NN / 16, 32, 0, stream>>>(x, Win16, b_in, lnin_g, lnin_b, hf32, h16);

  // ---- SAGE layers ----
  const size_t nh = (size_t)NN * HH;
  for (int l = 0; l < LLAY; ++l) {
    for (int r = 0; r < RREL; ++r) {
      zero_f32_kernel<<<(int)((nh + 255) / 256), 256, 0, stream>>>(agg, nh);
      zero_f32_kernel<<<(NN + 255) / 256, 256, 0, stream>>>((float*)cnt, NN);
      scatter_kernel<<<(EE * 32) / 256, 256, 0, stream>>>(
          edge_index + (size_t)r * 2 * EE, hf32, agg, cnt);
      int lr = l * RREL + r;
      sage_gemm_kernel<<<NN / 16, 32, 0, stream>>>(
          agg, cnt, h16, Wl16 + (size_t)lr * HH * HH, Wr16 + (size_t)lr * HH * HH,
          bl + (size_t)lr * HH, accb, (r == 0) ? 1 : 0);
    }
    if (l < LLAY - 1) {
      zero_f32_kernel<<<1, 256, 0, stream>>>(colsum, 256);  // colsum+colsq (contiguous)
      finalize0_kernel<<<NN / 32, 128, 0, stream>>>(accb, colsum, colsq);
      gn_finish_kernel<<<1, 128, 0, stream>>>(colsum, colsq, gn_w, gn_a, shiftb, mulb);
      gn_apply_kernel<<<(int)((nh + 255) / 256), 256, 0, stream>>>(accb, shiftb, mulb, gn_b, hf32, h16);
    } else {
      finalize_last_kernel<<<(int)((nh + 255) / 256), 256, 0, stream>>>(accb, out_h, h16);
    }
  }

  // ---- heads ----
  staff_kernel<<<NN / 16, 32, 0, stream>>>(h16, st16, st_b1, st_lng, st_lnb, st_W2, st_b2, out_staff);
  chord_kernel<<<ECH / 16, 32, 0, stream>>>(pot_chord, h16, cp16, cp_b1, cp_lng, cp_lnb,
                                            cp_W2, cp_b2, out_pool);
  decoder_kernel<<<EPOT / 16, 32, 0, stream>>>(pot_edges, h16, dd16, d_W1, d_b1, d_lng, d_lnb,
                                               d_W2, d_b2, onsets, durations, pitches,
                                               onset_beat, durat_beat, ts_beats, out_edge);
}
